// DeformableConv2D_91044716741198
// MI455X (gfx1250) — matrix operand order repaired and checked by reading, not compiled, not run on hardware
//
#include <hip/hip_runtime.h>
#include <hip/hip_bf16.h>

// ---------------------------------------------------------------------------
// Deformable Conv2D on gfx1250 (MI455X)
//  B=4, H=W=56, CIN=64, FILTERS=64, NDG=4, KS=3  -> K_taps=9, FIG=16
//  Stage 1: offset conv as implicit GEMM  [M=12544, K=576, N=72(->80)]  (WMMA bf16)
//  Stage 2+3: fused bilinear sampling + grouped GEMM [per group: M=12544,K=576,N=16]
// ---------------------------------------------------------------------------

typedef __attribute__((ext_vector_type(16))) __bf16          v16bf;
typedef __attribute__((ext_vector_type(8)))  float           v8f;
typedef __attribute__((ext_vector_type(8)))  unsigned short  u16x8;
typedef __attribute__((ext_vector_type(4)))  unsigned short  u16x4;
typedef __attribute__((ext_vector_type(4)))  float           f32x4;

#define BB   4
#define HH   56
#define WW   56
#define CIN  64
#define NDG  4
#define FIG  16
#define KTAP 9
#define KDIM 576          // 9*64 GEMM depth
#define MTOT 12544        // 4*56*56
#define NOFF 72           // K*NDG*2
#define NOFFP 80          // padded to 5 N-tiles
#define ASTRIDE 584       // 576 + 8 pad: 16B-aligned rows, conflict-free banks

__device__ __forceinline__ unsigned short f2bf(float f) {
    unsigned u = __float_as_uint(f);
    u += 0x7fffu + ((u >> 16) & 1u);   // round-to-nearest-even
    return (unsigned short)(u >> 16);
}

union BFVec { u16x8 h[2]; v16bf v; };

__device__ __forceinline__ v16bf load_frag(const unsigned short* p0,
                                           const unsigned short* p1) {
    BFVec u;
    u.h[0] = *(const u16x8*)p0;   // 16B load (ds_load_b128 / global_load_b128)
    u.h[1] = *(const u16x8*)p1;
    return u.v;
}

// ---------------------------------------------------------------------------
// Pack weights to bf16 in B-fragment layout: row n holds 576 consecutive K.
//   wo[(nt*16+n)*576 + k]  : offset-conv weights, k=(ki*3+kj)*64+cin, n<72 else 0
//   wm[((g*16)+r)*576 + k] : main-conv weights for group g, filter f=g*16+r
// ---------------------------------------------------------------------------
__global__ void pack_weights_kernel(const float* __restrict__ offk,
                                    const float* __restrict__ kern,
                                    unsigned short* __restrict__ wo,
                                    unsigned short* __restrict__ wm) {
    int idx = blockIdx.x * 256 + threadIdx.x;
    const int NW0 = NOFFP * KDIM;              // 46080
    const int NW1 = NDG * FIG * KDIM;          // 36864
    if (idx < NW0) {
        int n = idx / KDIM, k = idx % KDIM;
        float v = (n < NOFF) ? offk[(size_t)k * NOFF + n] : 0.0f;
        wo[idx] = f2bf(v);
    } else if (idx < NW0 + NW1) {
        int j = idx - NW0;
        int g = j / (FIG * KDIM);
        int r = (j / KDIM) & 15;
        int k = j % KDIM;
        int tap = k / CIN, cin = k % CIN;
        int f = g * FIG + r;
        wm[j] = f2bf(kern[((size_t)tap * 64 + f) * CIN + cin]);
    }
}

// ---------------------------------------------------------------------------
// Stage 1: offset conv. grid.x = 784 M-tiles; 5 waves = 5 N-tiles of 16.
// ---------------------------------------------------------------------------
__global__ __launch_bounds__(160) void offset_conv_kernel(
        const float* __restrict__ x,
        const unsigned short* __restrict__ wo,
        const float* __restrict__ bias,
        float* __restrict__ off_buf) {
    __shared__ unsigned short As[16][ASTRIDE];
    const int tid = threadIdx.x;
    const int m0  = blockIdx.x * 16;

    // Build im2col A tile (16 positions x 576 K) in bf16.
    for (int idx = tid; idx < 16 * KTAP * 4; idx += 160) {
        int p     = idx / 36;
        int rem   = idx % 36;
        int tap   = rem >> 2;
        int chunk = rem & 3;
        int m  = m0 + p;
        int b  = m / (HH * WW);
        int hw = m % (HH * WW);
        int h  = hw / WW, w = hw % WW;
        int ki = tap / 3, kj = tap % 3;
        int yy = h + ki - 1, xx = w + kj - 1;
        bool valid = (yy >= 0 && yy < HH && xx >= 0 && xx < WW);
        const float* src = x + (((size_t)b * HH * WW + (size_t)yy * WW + xx) * CIN
                                + chunk * 16);
        unsigned short* dst = &As[p][tap * CIN + chunk * 16];
        for (int cc = 0; cc < 4; ++cc) {
            f32x4 v = {0.f, 0.f, 0.f, 0.f};
            if (valid) v = *(const f32x4*)(src + cc * 4);
            u16x4 o;
            o.x = f2bf(v.x); o.y = f2bf(v.y); o.z = f2bf(v.z); o.w = f2bf(v.w);
            *(u16x4*)(dst + cc * 4) = o;
        }
    }
    __syncthreads();

    const int nt   = tid >> 5;        // N-tile = wave id (0..4)
    const int lane = tid & 31;
    const int mrow = lane & 15;
    const int hi   = lane >> 4;
    v8f acc = {0.f, 0.f, 0.f, 0.f, 0.f, 0.f, 0.f, 0.f};
    const unsigned short* wbase = wo + (size_t)(nt * 16 + (lane & 15)) * KDIM;

    for (int ks = 0; ks < KDIM / 32; ++ks) {
        int kb = ks * 32;
        // A fragment (16-bit A 16x32 layout): lane<16 -> K 0..7 | 16..23
        v16bf a = load_frag(&As[mrow][kb + hi * 8],
                            &As[mrow][kb + hi * 8 + 16]);
        // B fragment (16-bit B 32x16 layout): lane<16 -> K 0..7 | 16..23
        v16bf bm = load_frag(wbase + kb + hi * 8,
                             wbase + kb + hi * 8 + 16);
        acc = __builtin_amdgcn_wmma_f32_16x16x32_bf16(
                false, a, false, bm, (short)0, acc, false, false);
    }

    int ng = nt * 16 + (lane & 15);
    if (ng < NOFF) {
        float bv = bias[ng];
        for (int r = 0; r < 8; ++r) {
            int mr = r + hi * 8;
            off_buf[(size_t)(m0 + mr) * NOFF + ng] = acc[r] + bv;
        }
    }
}

// ---------------------------------------------------------------------------
// Stage 2+3: fused bilinear sampling + grouped GEMM.
// grid.x = 784 M-tiles; 4 waves = 4 deformable groups.
// ---------------------------------------------------------------------------
__global__ __launch_bounds__(128) void deform_gemm_kernel(
        const float* __restrict__ x,
        const float* __restrict__ off_buf,
        const unsigned short* __restrict__ wm,
        float* __restrict__ out) {
    __shared__ unsigned short As[NDG][16][ASTRIDE];   // ~73 KB
    const int tid = threadIdx.x;
    const int m0  = blockIdx.x * 16;

    // Cooperative bilinear sampling into per-group bf16 A tiles.
    for (int idx = tid; idx < 16 * 36 * 4; idx += 128) {
        int p     = idx / 144;
        int rem   = idx % 144;
        int tg    = rem >> 2;        // tap*NDG + g
        int chunk = rem & 3;
        int tap   = tg >> 2;
        int g     = tg & 3;
        int m  = m0 + p;
        int b  = m / (HH * WW);
        int hw = m % (HH * WW);
        int h  = hw / WW, w = hw % WW;
        int ki = tap / 3, kj = tap % 3;

        float yoff = off_buf[(size_t)m * NOFF + tg * 2];
        float xoff = off_buf[(size_t)m * NOFF + tg * 2 + 1];
        // padded coords in [0, 57]
        float yf = fminf(fmaxf((float)(h + ki) + yoff, 0.f), 57.f);
        float xf = fminf(fmaxf((float)(w + kj) + xoff, 0.f), 57.f);
        int y0 = (int)floorf(yf), x0 = (int)floorf(xf);
        int y1 = min(y0 + 1, 57), x1 = min(x0 + 1, 57);
        y0 = min(max(y0, 0), 57);   x0 = min(max(x0, 0), 57);
        float y0f = (float)y0, x0f = (float)x0;
        float y1f = (float)y1, x1f = (float)x1;
        float w0 = (y1f - yf) * (x1f - xf);
        float w1 = (y1f - yf) * (xf - x0f);
        float w2 = (yf - y0f) * (x1f - xf);
        float w3 = (yf - y0f) * (xf - x0f);

        const size_t bbase = (size_t)b * HH * WW * CIN;
        int c0 = chunk * 16;
        unsigned short* dst = &As[g][p][tap * CIN + c0];
        for (int cc = 0; cc < 4; ++cc) {
            int c = c0 + cc * 4;
            f32x4 z = {0.f, 0.f, 0.f, 0.f};
            f32x4 p00 = z, p01 = z, p10 = z, p11 = z;
            {   // gather 4 corners (padded coord -> unpadded, zero outside)
                int yu, xu;
                yu = y0 - 1; xu = x0 - 1;
                if (yu >= 0 && yu < HH && xu >= 0 && xu < WW)
                    p00 = *(const f32x4*)(x + bbase + ((size_t)yu * WW + xu) * CIN + c);
                yu = y0 - 1; xu = x1 - 1;
                if (yu >= 0 && yu < HH && xu >= 0 && xu < WW)
                    p01 = *(const f32x4*)(x + bbase + ((size_t)yu * WW + xu) * CIN + c);
                yu = y1 - 1; xu = x0 - 1;
                if (yu >= 0 && yu < HH && xu >= 0 && xu < WW)
                    p10 = *(const f32x4*)(x + bbase + ((size_t)yu * WW + xu) * CIN + c);
                yu = y1 - 1; xu = x1 - 1;
                if (yu >= 0 && yu < HH && xu >= 0 && xu < WW)
                    p11 = *(const f32x4*)(x + bbase + ((size_t)yu * WW + xu) * CIN + c);
            }
            f32x4 v = w0 * p00 + w1 * p01 + w2 * p10 + w3 * p11;
            u16x4 o;
            o.x = f2bf(v.x); o.y = f2bf(v.y); o.z = f2bf(v.z); o.w = f2bf(v.w);
            *(u16x4*)(dst + cc * 4) = o;
        }
    }
    __syncthreads();

    const int g    = tid >> 5;        // group = wave id
    const int lane = tid & 31;
    const int mrow = lane & 15;
    const int hi   = lane >> 4;
    v8f acc = {0.f, 0.f, 0.f, 0.f, 0.f, 0.f, 0.f, 0.f};
    const unsigned short* wbase = wm + ((size_t)g * FIG + (lane & 15)) * KDIM;

    for (int ks = 0; ks < KDIM / 32; ++ks) {
        int kb = ks * 32;
        v16bf a = load_frag(&As[g][mrow][kb + hi * 8],
                            &As[g][mrow][kb + hi * 8 + 16]);
        v16bf bm = load_frag(wbase + kb + hi * 8,
                             wbase + kb + hi * 8 + 16);
        acc = __builtin_amdgcn_wmma_f32_16x16x32_bf16(
                false, a, false, bm, (short)0, acc, false, false);
    }

    for (int r = 0; r < 8; ++r) {
        int mr = r + hi * 8;
        out[(size_t)(m0 + mr) * 64 + g * FIG + (lane & 15)] = acc[r];
    }
}

// ---------------------------------------------------------------------------
// Launch: pack -> offset conv -> fused deform GEMM   (workspace ~3.8 MB)
// ---------------------------------------------------------------------------
extern "C" void kernel_launch(void* const* d_in, const int* in_sizes, int n_in,
                              void* d_out, int out_size, void* d_ws, size_t ws_size,
                              hipStream_t stream) {
    const float* x    = (const float*)d_in[0];
    const float* offk = (const float*)d_in[1];
    const float* bias = (const float*)d_in[2];
    const float* kern = (const float*)d_in[3];
    float* out        = (float*)d_out;

    char* ws = (char*)d_ws;
    float*          off_buf = (float*)ws;                              // 12544*72 f32
    unsigned short* wo      = (unsigned short*)(ws + 3612672);         // 80*576 bf16
    unsigned short* wm      = (unsigned short*)(ws + 3612672 + 92160); // 64*576 bf16

    const int npack = NOFFP * KDIM + NDG * FIG * KDIM;   // 82944
    pack_weights_kernel<<<(npack + 255) / 256, 256, 0, stream>>>(offk, kern, wo, wm);
    offset_conv_kernel<<<MTOT / 16, 160, 0, stream>>>(x, wo, bias, off_buf);
    deform_gemm_kernel<<<MTOT / 16, 128, 0, stream>>>(x, off_buf, wm, out);

    (void)in_sizes; (void)n_in; (void)out_size; (void)ws_size;
}
